// myRNN_72310069395466
// MI455X (gfx1250) — matrix operand order repaired and checked by reading, not compiled, not run on hardware
//
#include <hip/hip_runtime.h>
#include <hip/hip_bf16.h>
#include <math.h>

typedef __attribute__((ext_vector_type(8)))  __bf16 v8bf;
typedef __attribute__((ext_vector_type(16))) __bf16 v16bf;
typedef __attribute__((ext_vector_type(8)))  float  v8f;

#define B_SZ   64
#define S_LEN  512
#define EMB_D  512
#define HID_D  1024
#define OUT_D  2
#define KTOT   3584   // 512 (Wih0) + 1024 (Whh0) + 1024 (Wih1) + 1024 (Whh1)
#define KPAD   3592   // +8 bf16 pad per column -> LDS bank stride 1796 dwords (%64 == 4)
#define NWG    64
#define KOFF_IH0 0
#define KOFF_HH0 512
#define KOFF_IH1 1536
#define KOFF_HH1 2560

// ---------- fragment helpers (layouts per CDNA5 ISA 7.12.2, wave32) ----------
static __device__ __forceinline__ v16bf join8(v8bf lo, v8bf hi) {
  v16bf r;
#pragma unroll
  for (int i = 0; i < 8; ++i) { r[i] = lo[i]; r[i + 8] = hi[i]; }
  return r;
}
// A-fragment (16-bit A 16x32): this lane's halves are K[sub..sub+7] and K[sub+16..sub+23]
static __device__ __forceinline__ v16bf load_a_bf16(const __bf16* p) {
  return join8(*(const v8bf*)p, *(const v8bf*)(p + 16));
}
// B-fragment: this lane's column halves are K[sub..sub+7] and K[sub+16..sub+23] (sub chosen by caller)
static __device__ __forceinline__ v16bf load_b16(const __bf16* p) {
  return join8(*(const v8bf*)p, *(const v8bf*)(p + 16));
}
static __device__ __forceinline__ v16bf cvt_f32x16(v8f f0, v8f f1) {
  v16bf r;
#pragma unroll
  for (int i = 0; i < 8; ++i) { r[i] = (__bf16)f0[i]; r[i + 8] = (__bf16)f1[i]; }
  return r;
}
static __device__ __forceinline__ v8f wmma_bf16(v16bf a, v16bf b, v8f c) {
  return __builtin_amdgcn_wmma_f32_16x16x32_bf16(false, a, false, b, (short)0, c,
                                                 false, false);
}

// bf16 A (hidden state row) x one B slice
template <int NC>
static __device__ __forceinline__ v8f mm_tile(const __bf16* arow, int sub,
                                              const __bf16* bcol, int koff, v8f acc) {
#pragma unroll 4
  for (int c = 0; c < NC; ++c) {
    v16bf a = load_a_bf16(arow + c * 32 + sub);
    acc = wmma_bf16(a, load_b16(bcol + koff + c * 32), acc);
  }
  return acc;
}
// Fused phase: three INDEPENDENT WMMA chains per K-chunk
//   acc1a += h0(t) @ Wih1     (layer-1 input term)
//   acc0  += h0(t) @ Whh0     (next-step layer-0 recurrent term; shares A with above)
//   acc1b += h1(t-1) @ Whh1   (layer-1 recurrent term)
template <int NC>
static __device__ __forceinline__ void mm_phase3(const __bf16* a0row,
                                                 const __bf16* a1row, int sub,
                                                 const __bf16* bcol,
                                                 v8f& acc1a, v8f& acc1b, v8f& acc0) {
#pragma unroll 2
  for (int c = 0; c < NC; ++c) {
    v16bf a0 = load_a_bf16(a0row + c * 32 + sub);
    v16bf a1 = load_a_bf16(a1row + c * 32 + sub);
    acc1a = wmma_bf16(a0, load_b16(bcol + KOFF_IH1 + c * 32), acc1a);
    acc0  = wmma_bf16(a0, load_b16(bcol + KOFF_HH0 + c * 32), acc0);
    acc1b = wmma_bf16(a1, load_b16(bcol + KOFF_HH1 + c * 32), acc1b);
  }
}
// fp32 embedding-row A operand, converted on the fly (fallback path)
static __device__ __forceinline__ v8f mm_emb(const float* er, int sub,
                                             const __bf16* bcol, v8f acc) {
#pragma unroll 4
  for (int c = 0; c < EMB_D / 32; ++c) {
    v8f f0 = *(const v8f*)(er + c * 32 + sub);
    v8f f1 = *(const v8f*)(er + c * 32 + sub + 16);
    v16bf a = cvt_f32x16(f0, f1);
    acc = wmma_bf16(a, load_b16(bcol + KOFF_IH0 + c * 32), acc);
  }
  return acc;
}
// D-fragment (f32 16x16): lane holds col = lane&15, rows = vgpr + 8*(lane>>4)
static __device__ __forceinline__ void store_h(v8f acc, __bf16* h, int mrow0, int colg) {
#pragma unroll
  for (int i = 0; i < 8; ++i) {
    h[(size_t)(mrow0 + i) * HID_D + colg] = (__bf16)tanhf(acc[i]);
  }
}
static __device__ __forceinline__ size_t z0_off(int t, int nt, int m, int lane) {
  return ((((size_t)t * NWG + nt) * 4 + m) * 32 + lane) * 8;
}

// ---------- grid-wide barrier (agent-scope atomics, generation counter) ----------
static __device__ __forceinline__ void grid_sync(unsigned* counter, unsigned* gen,
                                                 unsigned nwg) {
  __syncthreads();
  if (threadIdx.x == 0) {
    unsigned g = __hip_atomic_load(gen, __ATOMIC_RELAXED, __HIP_MEMORY_SCOPE_AGENT);
    unsigned old =
        __hip_atomic_fetch_add(counter, 1u, __ATOMIC_ACQ_REL, __HIP_MEMORY_SCOPE_AGENT);
    if (old == nwg - 1u) {
      __hip_atomic_store(counter, 0u, __ATOMIC_RELAXED, __HIP_MEMORY_SCOPE_AGENT);
      __hip_atomic_fetch_add(gen, 1u, __ATOMIC_ACQ_REL, __HIP_MEMORY_SCOPE_AGENT);
    } else {
      while (__hip_atomic_load(gen, __ATOMIC_ACQUIRE, __HIP_MEMORY_SCOPE_AGENT) == g) {
        __builtin_amdgcn_s_sleep(1);
      }
    }
  }
  __syncthreads();
}

// ---------- pre-pass 1: bf16-convert + transpose all 4 weight matrices ----------
// WT layout: [n (1024)][KTOT] with K = concat(Wih0^T, Whh0^T, Wih1^T, Whh1^T)
__global__ void convert_weights(const float* __restrict__ Wih0,
                                const float* __restrict__ Whh0,
                                const float* __restrict__ Wih1,
                                const float* __restrict__ Whh1,
                                __bf16* __restrict__ WT) {
  const int n = blockIdx.x;
  for (int j = threadIdx.x; j < KTOT; j += blockDim.x) {
    float v;
    if (j < KOFF_HH0)      v = Wih0[(size_t)j * HID_D + n];
    else if (j < KOFF_IH1) v = Whh0[(size_t)(j - KOFF_HH0) * HID_D + n];
    else if (j < KOFF_HH1) v = Wih1[(size_t)(j - KOFF_IH1) * HID_D + n];
    else                   v = Whh1[(size_t)(j - KOFF_HH1) * HID_D + n];
    WT[(size_t)n * KTOT + j] = (__bf16)v;
  }
}

// ---------- pre-pass 2: Z0[t] = X_t @ Wih0 for all t (off the serial chain) ----
// Output stored in D-fragment layout: [t][nt][m][lane][8 floats]
__global__ void emb_proj(const int* __restrict__ x, const float* __restrict__ emb,
                         const __bf16* __restrict__ WT, float* __restrict__ Z0) {
  const int t    = blockIdx.x;
  const int g    = blockIdx.y * 8 + (threadIdx.x >> 5);  // tile id 0..255
  const int nt   = g >> 2;
  const int m    = g & 3;
  const int lane = threadIdx.x & 31;
  const int lo16 = lane & 15;
  const int hi   = lane >> 4;
  const int sub  = hi * 8;
  const int row  = m * 16 + lo16;
  const int tok  = x[(size_t)row * S_LEN + t];
  const float* er = emb + (size_t)tok * EMB_D;
  const __bf16* bcol = WT + (size_t)(nt * 16 + lo16) * KTOT + hi * 8;
  v8f acc;
#pragma unroll
  for (int i = 0; i < 8; ++i) acc[i] = 0.0f;
#pragma unroll 4
  for (int c = 0; c < EMB_D / 32; ++c) {
    v8f f0 = *(const v8f*)(er + c * 32 + sub);
    v8f f1 = *(const v8f*)(er + c * 32 + sub + 16);
    acc = wmma_bf16(cvt_f32x16(f0, f1), load_b16(bcol + c * 32), acc);
  }
  *(v8f*)(Z0 + z0_off(t, nt, m, lane)) = acc;
}

// ---------- persistent RNN: 64 WGs x 128 threads, ONE grid sync per step ------
// Phase structure: A(0) | sync | {L1(0), L0(1)} | sync | {L1(1), L0(2)} | ...
__global__ void rnn_persistent(const int* __restrict__ x, const float* __restrict__ emb,
                               const float* __restrict__ bh0,
                               const float* __restrict__ bh1,
                               const __bf16* __restrict__ WT,
                               const float* __restrict__ Z0, int use_z0,
                               __bf16* h0A, __bf16* h0B, __bf16* h1A, __bf16* h1B,
                               unsigned* counter, unsigned* gen) {
  extern __shared__ __bf16 lds[];  // [16 cols][KPAD]
  const int nt   = blockIdx.x;
  const int m    = threadIdx.x >> 5;
  const int lane = threadIdx.x & 31;
  const int lo16 = lane & 15;
  const int hi   = lane >> 4;
  const unsigned nwg = gridDim.x;

  // one-time preload of this WG's 16-column weight slice into LDS (112 KB)
  for (int idx = threadIdx.x; idx < 16 * (KTOT / 8); idx += blockDim.x) {
    const int j = idx / (KTOT / 8);
    const int c = idx % (KTOT / 8);
    *(v8bf*)(lds + j * KPAD + c * 8) =
        *(const v8bf*)(WT + (size_t)(16 * nt + j) * KTOT + c * 8);
  }
  __syncthreads();

  const int   colg  = nt * 16 + lo16;
  const float bias0 = bh0[colg];
  const float bias1 = bh1[colg];
  const int   row   = m * 16 + lo16;
  const int   sub   = hi * 8;
  const int   mrow0 = m * 16 + hi * 8;
  const __bf16* bcol = lds + lo16 * KPAD + hi * 8;

  // prologue: h0(0) = tanh(x0@Wih0 + 0@Whh0 + b0)   (h0A holds zeros)
  {
    v8f acc0;
#pragma unroll
    for (int i = 0; i < 8; ++i) acc0[i] = bias0;
    if (use_z0) {
      v8f z = *(const v8f*)(Z0 + z0_off(0, nt, m, lane));
#pragma unroll
      for (int i = 0; i < 8; ++i) acc0[i] += z[i];
    } else {
      const int tok = x[(size_t)row * S_LEN];
      acc0 = mm_emb(emb + (size_t)tok * EMB_D, sub, bcol, acc0);
    }
    acc0 = mm_tile<HID_D / 32>(h0A + (size_t)row * HID_D, sub, bcol, KOFF_HH0, acc0);
    store_h(acc0, h0B, mrow0, colg);
  }
  grid_sync(counter, gen, nwg);

  __bf16* h0c = h0B;  __bf16* h0a = h0A;   // h0c = h0(t); h0a = scratch for h0(t+1)
  __bf16* h1p = h1A;  __bf16* h1c = h1B;   // h1p = h1(t-1) (zeros initially)

  for (int t = 0; t < S_LEN; ++t) {
    const int tn = (t + 1 < S_LEN) ? t + 1 : S_LEN - 1;  // clamped (no OOB on last)

    if (use_z0 && t + 2 < S_LEN)
      __builtin_prefetch(Z0 + z0_off(t + 2, nt, m, lane), 0, 1);

    // acc1a+acc1b -> h1(t) pre-activation ; acc0 -> h0(t+1) pre-activation
    v8f acc1a, acc1b, acc0;
#pragma unroll
    for (int i = 0; i < 8; ++i) acc1a[i] = bias1;
#pragma unroll
    for (int i = 0; i < 8; ++i) acc1b[i] = 0.0f;
#pragma unroll
    for (int i = 0; i < 8; ++i) acc0[i] = bias0;

    if (use_z0) {
      v8f z = *(const v8f*)(Z0 + z0_off(tn, nt, m, lane));
#pragma unroll
      for (int i = 0; i < 8; ++i) acc0[i] += z[i];
    } else {
      const int tok = x[(size_t)row * S_LEN + tn];
      acc0 = mm_emb(emb + (size_t)tok * EMB_D, sub, bcol, acc0);
    }

    // three independent WMMA chains over the shared K loop
    mm_phase3<HID_D / 32>(h0c + (size_t)row * HID_D, h1p + (size_t)row * HID_D,
                          sub, bcol, acc1a, acc1b, acc0);

#pragma unroll
    for (int i = 0; i < 8; ++i) acc1a[i] += acc1b[i];

    store_h(acc1a, h1c, mrow0, colg);
    if (t + 1 < S_LEN) store_h(acc0, h0a, mrow0, colg);

    grid_sync(counter, gen, nwg);

    __bf16* tmp;
    tmp = h0c; h0c = h0a; h0a = tmp;
    tmp = h1p; h1p = h1c; h1c = tmp;
  }
}

// ---------- epilogue: out = h1_T @ W_out + b_out  ([64x1024]x[1024x2]) ----------
__global__ void out_proj(const __bf16* __restrict__ h1, const float* __restrict__ Wout,
                         const float* __restrict__ bout, float* __restrict__ out) {
  const int tid = blockIdx.x * blockDim.x + threadIdx.x;
  if (tid >= B_SZ * OUT_D) return;
  const int b = tid >> 1, o = tid & 1;
  float s = bout[o];
  for (int k = 0; k < HID_D; ++k)
    s += (float)h1[(size_t)b * HID_D + k] * Wout[(size_t)k * OUT_D + o];
  out[tid] = s;
}

extern "C" void kernel_launch(void* const* d_in, const int* in_sizes, int n_in,
                              void* d_out, int out_size, void* d_ws, size_t ws_size,
                              hipStream_t stream) {
  (void)in_sizes; (void)n_in; (void)out_size;
  const int*   x    = (const int*)  d_in[0];
  const float* emb  = (const float*)d_in[1];
  const float* Wih0 = (const float*)d_in[2];
  const float* Whh0 = (const float*)d_in[3];
  const float* bh0  = (const float*)d_in[4];
  const float* Wih1 = (const float*)d_in[5];
  const float* Whh1 = (const float*)d_in[6];
  const float* bh1  = (const float*)d_in[7];
  const float* Wout = (const float*)d_in[8];
  const float* bout = (const float*)d_in[9];
  float* out = (float*)d_out;

  // workspace carve-up (256B aligned)
  char* ws = (char*)d_ws;
  __bf16* WT = (__bf16*)ws;                               // 7,340,032 B
  size_t off = ((size_t)HID_D * KTOT * 2 + 255) & ~(size_t)255;
  const size_t hbytes = (size_t)B_SZ * HID_D * 2;         // 131,072 B each
  __bf16* h0A = (__bf16*)(ws + off);
  __bf16* h0B = (__bf16*)(ws + off + hbytes);
  __bf16* h1A = (__bf16*)(ws + off + 2 * hbytes);
  __bf16* h1B = (__bf16*)(ws + off + 3 * hbytes);
  unsigned* syncp = (unsigned*)(ws + off + 4 * hbytes);   // [0]=counter, [1]=gen
  size_t zoff = (off + 4 * hbytes + 256 + 255) & ~(size_t)255;
  float* Z0 = (float*)(ws + zoff);
  const size_t z0_bytes = (size_t)S_LEN * NWG * 4 * 32 * 8 * sizeof(float); // 134 MB
  const int use_z0 = (ws_size >= zoff + z0_bytes) ? 1 : 0;

  hipMemsetAsync(h0A, 0, 4 * hbytes, stream);             // zero h state
  hipMemsetAsync(syncp, 0, 256, stream);                  // zero barrier state

  convert_weights<<<HID_D, 256, 0, stream>>>(Wih0, Whh0, Wih1, Whh1, WT);
  if (use_z0)
    emb_proj<<<dim3(S_LEN, 32), 256, 0, stream>>>(x, emb, WT, Z0);

  const size_t lds_bytes = (size_t)16 * KPAD * sizeof(__bf16);  // 114,944 B
  rnn_persistent<<<NWG, 128, lds_bytes, stream>>>(x, emb, bh0, bh1, WT, Z0, use_z0,
                                                  h0A, h0B, h1A, h1B,
                                                  syncp, syncp + 1);

  // h1 writes alternate B(t even)/A(t odd); final t=511 -> h1A
  out_proj<<<1, 128, 0, stream>>>(h1A, Wout, bout, out);
}
